// RGAT_36309653521093
// MI455X (gfx1250) — compile-verified, matrix operand order checked by reading, not run on hardware
//
#include <hip/hip_runtime.h>

// ---------------- CDNA5 types ----------------
typedef __attribute__((ext_vector_type(16))) __bf16 v16bf;
typedef __attribute__((ext_vector_type(8)))  float  v8f;
typedef __attribute__((ext_vector_type(4)))  unsigned int u32x4;
typedef __attribute__((ext_vector_type(2)))  unsigned int u32x2;
typedef __attribute__((ext_vector_type(4)))  float  f32x4;

// ---------------- problem constants ----------------
static constexpr int CH   = 128;   // channels
static constexpr int NJ   = 17;    // joints per graph
static constexpr int NE   = 64;    // directed edges per graph
static constexpr int GPB  = 8;     // graphs per block
static constexpr int ROWS = GPB * NJ;       // 136 real rows
static constexpr int MP   = 144;            // padded rows: 9 tiles of 16
static constexpr int KP   = 136;            // bf16 row stride (272B, 16B aligned, 4-bank rotate)
static constexpr int X0P  = 132;            // fp32 row stride for x0 (528B rows, 16B aligned)
static constexpr int NT   = 256;            // threads (8 waves)

// ---------------- LDS layout (bytes) ----------------
static constexpr size_t OFF_WL   = 0;
static constexpr size_t OFF_WR   = OFF_WL + (size_t)CH * KP * 2;
static constexpr size_t OFF_A    = OFF_WR + (size_t)CH * KP * 2;
static constexpr size_t OFF_B    = OFF_A + (size_t)MP * KP * 2;
static constexpr size_t OFF_C    = OFF_B + (size_t)MP * KP * 2;
static constexpr size_t OFF_X0   = OFF_C + (size_t)MP * KP * 2;
static constexpr size_t OFF_VEC  = OFF_X0 + (size_t)ROWS * X0P * 4;  // 6 vectors of CH floats
static constexpr size_t OFF_LOG  = OFF_VEC + 6 * (size_t)CH * 4;     // GPB*NE floats
static constexpr size_t OFF_MAX  = OFF_LOG + (size_t)GPB * NE * 4;   // ROWS floats
static constexpr size_t OFF_DEN  = OFF_MAX + (size_t)ROWS * 4;       // ROWS floats
static constexpr size_t OFF_INT  = OFF_DEN + (size_t)ROWS * 4;       // src,dst,off,idx
static constexpr size_t SMEM_BYTES = OFF_INT + (2 * NE + (NJ + 1) + NE) * 4;

// ---------------- bf16 GEMM tile: dst = act @ W^T(+bias), all in LDS ----------------
__device__ __forceinline__ void lds_gemm(const __bf16* act, const __bf16* wt,
                                         const float* bias, __bf16* dst,
                                         int wave, int lane) {
  const int half  = lane >> 4;    // K-half selector per ISA A/B layout
  const int lan16 = lane & 15;
  for (int t = wave; t < (MP / 16) * 8; t += 8) {
    const int mt = t >> 3;                       // 0..8
    const int nt = t & 7;                        // 0..7
    const int n  = nt * 16 + lan16;
    const float bv = bias[n];
    v8f acc;
#pragma unroll
    for (int v = 0; v < 8; ++v) acc[v] = bv;     // C = broadcast bias (D-layout column n)
    const __bf16* arow = act + (mt * 16 + lan16) * KP + half * 8;
    const __bf16* brow = wt + n * KP + half * 8;
#pragma unroll
    for (int ks = 0; ks < 4; ++ks) {
      union { u32x4 u[2]; v16bf v; } af, bf;
      af.u[0] = *(const u32x4*)(arow + ks * 32);
      af.u[1] = *(const u32x4*)(arow + ks * 32 + 16);
      bf.u[0] = *(const u32x4*)(brow + ks * 32);
      bf.u[1] = *(const u32x4*)(brow + ks * 32 + 16);
      acc = __builtin_amdgcn_wmma_f32_16x16x32_bf16(
          false, af.v, false, bf.v, (short)0, acc, false, false);
    }
#pragma unroll
    for (int v = 0; v < 8; ++v) {
      const int row = mt * 16 + half * 8 + v;    // D-layout: M = v + 8*(lane>=16)
      dst[row * KP + n] = (__bf16)acc[v];
    }
  }
}

extern "C" __global__ void __launch_bounds__(NT)
rgat_fused(const float* __restrict__ x, const float* __restrict__ gamma,
           const float* __restrict__ beta, const float* __restrict__ alpha,
           const float* __restrict__ Wl, const float* __restrict__ bl,
           const float* __restrict__ Wr, const float* __restrict__ br,
           const float* __restrict__ att, const float* __restrict__ bias,
           const int* __restrict__ esrc, const int* __restrict__ edst,
           float* __restrict__ out) {
  extern __shared__ char smem[];
  __bf16* sWl  = (__bf16*)(smem + OFF_WL);
  __bf16* sWr  = (__bf16*)(smem + OFF_WR);
  __bf16* bufA = (__bf16*)(smem + OFF_A);   // h (bf16)
  __bf16* bufB = (__bf16*)(smem + OFF_B);   // xl
  __bf16* bufC = (__bf16*)(smem + OFF_C);   // xr
  float*  sx0  = (float*)(smem + OFF_X0);
  float*  svec = (float*)(smem + OFF_VEC);
  float*  sbl = svec, *sbr = svec + CH, *sbias = svec + 2 * CH,
       *satt = svec + 3 * CH, *sg = svec + 4 * CH, *sb = svec + 5 * CH;
  float* slog = (float*)(smem + OFF_LOG);
  float* smax = (float*)(smem + OFF_MAX);
  float* sden = (float*)(smem + OFF_DEN);
  int* ssrc = (int*)(smem + OFF_INT);
  int* sdst = ssrc + NE;
  int* soff = sdst + NE;
  int* sidx = soff + (NJ + 1);

  const int tid  = threadIdx.x, lane = tid & 31;
  const int wave = __builtin_amdgcn_readfirstlane(tid >> 5);  // uniform -> scalar loop
  const size_t base = (size_t)blockIdx.x * ROWS * CH;

  // ---- phase 0: weights (transposed -> bf16, b128 global loads), vectors, edges ----
  for (int i = tid; i < (CH * CH) / 4; i += NT) {
    const int k = i >> 5, n0 = (i & 31) * 4;    // W row-major [k][n0..n0+3]
    const f32x4 wl4 = *(const f32x4*)(Wl + i * 4);
    const f32x4 wr4 = *(const f32x4*)(Wr + i * 4);
#pragma unroll
    for (int q = 0; q < 4; ++q) {
      sWl[(n0 + q) * KP + k] = (__bf16)wl4[q];
      sWr[(n0 + q) * KP + k] = (__bf16)wr4[q];
    }
  }
  for (int i = tid; i < CH; i += NT) {
    sbl[i] = bl[i]; sbr[i] = br[i]; sbias[i] = bias[i];
    satt[i] = att[i]; sg[i] = gamma[i]; sb[i] = beta[i];
  }
  if (tid < NE) { ssrc[tid] = esrc[tid] % NJ; sdst[tid] = edst[tid] % NJ; }
  for (int i = tid; i < (MP - ROWS) * KP; i += NT)   // zero M-padding rows of h
    bufA[ROWS * KP + i] = (__bf16)0.0f;
  __syncthreads();
  if (tid == 0) {                                    // tiny CSR over 64 local edges
    int deg[NJ];
    for (int j = 0; j < NJ; ++j) deg[j] = 0;
    for (int e = 0; e < NE; ++e) deg[sdst[e]]++;
    int a = 0;
    for (int j = 0; j < NJ; ++j) { soff[j] = a; a += deg[j]; }
    soff[NJ] = a;
    int cur[NJ];
    for (int j = 0; j < NJ; ++j) cur[j] = soff[j];
    for (int e = 0; e < NE; ++e) sidx[cur[sdst[e]]++] = e;
  }

  // ---- phase 1: b128 coalesced tile load -> x0, then LayerNorm -> h (bf16) ----
  for (int i = tid; i < (ROWS * CH) / 4; i += NT) {
    const int r = i >> 5, c = (i & 31) * 4;
    *(f32x4*)(sx0 + r * X0P + c) = *(const f32x4*)(x + base + i * 4);
  }
  __syncthreads();
  for (int r = tid; r < ROWS; r += NT) {
    const float* xr = sx0 + r * X0P;
    float mu = 0.f;
    for (int c = 0; c < CH; ++c) mu += xr[c];
    mu *= (1.0f / CH);
    float var = 0.f;
    for (int c = 0; c < CH; ++c) { const float d = xr[c] - mu; var += d * d; }
    const float rs = rsqrtf(var * (1.0f / CH) + 1e-5f);
    for (int c = 0; c < CH; ++c)
      bufA[r * KP + c] = (__bf16)((xr[c] - mu) * rs * sg[c] + sb[c]);
  }
  const float s = 1.0f / (1.0f + __expf(-alpha[0]));
  __syncthreads();

  // ---- 4 fused GATv2 layers, all graph-local in LDS ----
  for (int layer = 0; layer < 4; ++layer) {
    lds_gemm(bufA, sWl, sbl, bufB, wave, lane);   // xl = h@Wl + bl
    lds_gemm(bufA, sWr, sbr, bufC, wave, lane);   // xr = h@Wr + br
    __syncthreads();

    // per-edge logits: att . leaky(xl[src] + xr[dst])
    for (int e = tid; e < GPB * NE; e += NT) {
      const int g = e >> 6, el = e & 63;
      const __bf16* pl = bufB + (g * NJ + ssrc[el]) * KP;
      const __bf16* pr = bufC + (g * NJ + sdst[el]) * KP;
      float sum = 0.f;
      for (int c8 = 0; c8 < 16; ++c8) {
        union { u32x4 u; __bf16 h[8]; } ul, ur;
        ul.u = *(const u32x4*)(pl + c8 * 8);
        ur.u = *(const u32x4*)(pr + c8 * 8);
#pragma unroll
        for (int q = 0; q < 8; ++q) {
          float v = (float)ul.h[q] + (float)ur.h[q];
          v = v > 0.f ? v : 0.2f * v;
          sum += satt[c8 * 8 + q] * v;
        }
      }
      slog[e] = sum;
    }
    __syncthreads();

    // segment-softmax stats per destination node
    for (int t = tid; t < ROWS; t += NT) {
      const int g = t / NJ, j = t % NJ;
      const float* lg = slog + g * NE;
      float m = -1e30f;
      for (int p = soff[j]; p < soff[j + 1]; ++p) m = fmaxf(m, lg[sidx[p]]);
      float den = 0.f;
      for (int p = soff[j]; p < soff[j + 1]; ++p) den += __expf(lg[sidx[p]] - m);
      smax[t] = m;
      sden[t] = 1.0f / den;
    }
    __syncthreads();
    for (int e = tid; e < GPB * NE; e += NT) {     // a[e] in place of logits
      const int g = e >> 6, dn = g * NJ + sdst[e & 63];
      slog[e] = __expf(slog[e] - smax[dn]) * sden[dn];
    }
    __syncthreads();

    // aggregate: h_new[dst] = sum a*xl[src] + bias, then layer combine
    for (int it = tid; it < ROWS * 16; it += NT) {
      const int r = it >> 4, c8 = it & 15;
      const int g = r / NJ, j = r % NJ;
      float acc[8];
#pragma unroll
      for (int q = 0; q < 8; ++q) acc[q] = sbias[c8 * 8 + q];
      for (int p = soff[j]; p < soff[j + 1]; ++p) {
        const int el = sidx[p];
        const float a = slog[g * NE + el];
        union { u32x4 u; __bf16 h[8]; } ux;
        ux.u = *(const u32x4*)(bufB + (g * NJ + ssrc[el]) * KP + c8 * 8);
#pragma unroll
        for (int q = 0; q < 8; ++q) acc[q] += a * (float)ux.h[q];
      }
      union { u32x4 u; __bf16 h[8]; } uo;
#pragma unroll
      for (int q = 0; q < 8; ++q) {
        float hv = acc[q];
        if (layer > 0) hv = (1.0f - s) * hv + s * sx0[r * X0P + c8 * 8 + q];
        uo.h[q] = (__bf16)hv;
      }
      *(u32x4*)(bufA + r * KP + c8 * 8) = uo.u;    // overwrite old h (GEMMs done)
    }
    __syncthreads();
  }

  // ---- epilogue: out = xf + h, b128 stores ----
  for (int i = tid; i < (ROWS * CH) / 4; i += NT) {
    const int r = i >> 5, c = (i & 31) * 4;
    const f32x4 x4 = *(const f32x4*)(sx0 + r * X0P + c);
    union { u32x2 u; __bf16 h[4]; } uh;
    uh.u = *(const u32x2*)(bufA + r * KP + c);
    f32x4 o4;
#pragma unroll
    for (int q = 0; q < 4; ++q) o4[q] = x4[q] + (float)uh.h[q];
    *(f32x4*)(out + base + i * 4) = o4;
  }
}

extern "C" void kernel_launch(void* const* d_in, const int* in_sizes, int n_in,
                              void* d_out, int out_size, void* d_ws, size_t ws_size,
                              hipStream_t stream) {
  const float* x     = (const float*)d_in[0];
  const float* gamma = (const float*)d_in[1];
  const float* beta  = (const float*)d_in[2];
  const float* alpha = (const float*)d_in[3];
  const float* Wl    = (const float*)d_in[4];
  const float* bl    = (const float*)d_in[5];
  const float* Wr    = (const float*)d_in[6];
  const float* br    = (const float*)d_in[7];
  const float* att   = (const float*)d_in[8];
  const float* bias  = (const float*)d_in[9];
  const int* esrc    = (const int*)d_in[10];
  const int* edst    = (const int*)d_in[11];

  const int nodes   = in_sizes[0] / CH;       // 264384
  const int nblocks = nodes / ROWS;           // 1944 (exact: 15552 graphs / 8)

  (void)hipFuncSetAttribute((const void*)rgat_fused,
                            hipFuncAttributeMaxDynamicSharedMemorySize,
                            (int)SMEM_BYTES);
  rgat_fused<<<dim3(nblocks), dim3(NT), SMEM_BYTES, stream>>>(
      x, gamma, beta, alpha, Wl, bl, Wr, br, att, bias, esrc, edst,
      (float*)d_out);
}
